// DynamicRouting_30562987278754
// MI455X (gfx1250) — matrix operand order repaired and checked by reading, not compiled, not run on hardware
//
#include <hip/hip_runtime.h>
#include <hip/hip_bf16.h>

// ---------------------------------------------------------------------------
// Fused grouped-1x1-conv + dynamic routing for MI455X (gfx1250, wave32, WMMA)
//
// con[pix, g, f] = sum_k x[pix, g*64+k] * kernels[g, k, f]      (8 GEMMs, K=64)
// Routing (3 iters) collapses algebraically to:
//   t  = sum_g con_g                      beta1_g = 0.5 * <con_g, t>_f
//   v1 = sum_g sigmoid(beta1_g) con_g     beta2_g = beta1_g + <con_g, v1>_f
//   out = sum_g sigmoid(beta2_g) con_g + bias
// con never touches HBM: each wave keeps its group's 16x256 f32 tile in VGPRs.
// ---------------------------------------------------------------------------

typedef __attribute__((ext_vector_type(16))) __bf16 v16bf;
typedef __attribute__((ext_vector_type(8)))  float  v8f;

__device__ __forceinline__ unsigned short f32_to_bf16_rne(float f) {
    union { float f; unsigned u; } c; c.f = f;
    unsigned u = c.u;
    unsigned r = u + 0x7FFFu + ((u >> 16) & 1u);   // round-to-nearest-even
    return (unsigned short)(r >> 16);
}

__device__ __forceinline__ float sigmoidf(float x) {
    return 1.0f / (1.0f + __expf(-x));
}

// ---------------------------------------------------------------------------
// Kernel 1: repack kernels fp32 [8][64][256] -> bf16 WMMA B-fragments in d_ws.
// Fragment convention for 16-bit B (32x16, K x N), one v16bf per lane:
//   lane L: N = L&15 ; i<8: K = i + 8*(L>>4) ; i>=8: K = i + 8 + 8*(L>>4)
// Storage: frag[g][kt][nf][lane][i], 2B elements -> main kernel loads each
// lane's fragment as one aligned 32B chunk (2 x global_load_b128).
// ---------------------------------------------------------------------------
__global__ __launch_bounds__(256)
void repack_kernels_bf16(const float* __restrict__ kern,
                         unsigned short* __restrict__ frag) {
    int o = blockIdx.x * 256 + threadIdx.x;        // 0 .. 131071
    int i    =  o        & 15;
    int lane = (o >> 4)  & 31;
    int nf   = (o >> 9)  & 15;
    int kt   = (o >> 13) & 1;
    int g    =  o >> 14;
    int k = kt * 32 + ((i & 8) * 2 + (lane >> 4) * 8 + (i & 7));      // 0..63
    int f = nf * 16 + (lane & 15);                 // 0..255
    frag[o] = f32_to_bf16_rne(kern[(g * 64 + k) * 256 + f]);
}

// ---------------------------------------------------------------------------
// Kernel 2: fused GEMM + routing. 256 threads = 8 waves, wave w <-> group g=w.
// M-tile = 16 pixels per block. LDS: two padded 16x260 f32 accumulators
// (row stride 260 makes the two wave-halves hit disjoint bank sets).
// ---------------------------------------------------------------------------
#define ROWS 16
#define RS   260

__global__ __launch_bounds__(256)
void routed_group_gemm(const float* __restrict__ x,
                       const unsigned short* __restrict__ bfrag,
                       const float* __restrict__ bias,
                       float* __restrict__ out) {
    __shared__ float bufA[ROWS * RS];
    __shared__ float bufB[ROWS * RS];

    const int tid  = threadIdx.x;
    const int g    = tid >> 5;            // wave id == group id
    const int lane = tid & 31;
    const int h    = lane >> 4;           // wave half (C-layout: m = r + 8h)
    const int fo   = lane & 15;           // f within 16-wide N tile
    const long long pixBase = (long long)blockIdx.x * ROWS;

    // zero both LDS accumulators (barrier deferred until after GEMM)
    for (int i = tid; i < ROWS * RS; i += 256) { bufA[i] = 0.0f; bufB[i] = 0.0f; }

    // ---- A fragments: x[16 pixels, 64 ch of group g] -> bf16, 2 K-tiles ----
    // 16-bit A 16x32 layout: lane L, m = L&15; i<8: K = i + 8h ; i>=8: K = i+8+8h
    union { v16bf v; unsigned short u[16]; } afrag[2];
    {
        const float* row = x + (pixBase + fo) * 512 + g * 64;
        #pragma unroll
        for (int kt = 0; kt < 2; ++kt) {
            const float* p = row + kt * 32 + 8 * h;
            float4 q0 = *(const float4*)(p + 0);    // K = 8h + 0..3
            float4 q1 = *(const float4*)(p + 4);    // K = 8h + 4..7
            float4 q2 = *(const float4*)(p + 16);   // K = 16 + 8h + 0..3
            float4 q3 = *(const float4*)(p + 20);   // K = 16 + 8h + 4..7
            afrag[kt].u[0]  = f32_to_bf16_rne(q0.x); afrag[kt].u[1]  = f32_to_bf16_rne(q0.y);
            afrag[kt].u[2]  = f32_to_bf16_rne(q0.z); afrag[kt].u[3]  = f32_to_bf16_rne(q0.w);
            afrag[kt].u[4]  = f32_to_bf16_rne(q1.x); afrag[kt].u[5]  = f32_to_bf16_rne(q1.y);
            afrag[kt].u[6]  = f32_to_bf16_rne(q1.z); afrag[kt].u[7]  = f32_to_bf16_rne(q1.w);
            afrag[kt].u[8]  = f32_to_bf16_rne(q2.x); afrag[kt].u[9]  = f32_to_bf16_rne(q2.y);
            afrag[kt].u[10] = f32_to_bf16_rne(q2.z); afrag[kt].u[11] = f32_to_bf16_rne(q2.w);
            afrag[kt].u[12] = f32_to_bf16_rne(q3.x); afrag[kt].u[13] = f32_to_bf16_rne(q3.y);
            afrag[kt].u[14] = f32_to_bf16_rne(q3.z); afrag[kt].u[15] = f32_to_bf16_rne(q3.w);
        }
    }

    // ---- GEMM: con_g[16 pixels, 256 f] in 16 v8f accumulators (128 VGPRs) --
    v8f acc[16] = {};
    const unsigned short* bg = bfrag + (size_t)g * 2 * 16 * 32 * 16;
    #pragma unroll
    for (int nf = 0; nf < 16; ++nf) {
        v16bf b0 = *(const v16bf*)(bg + ((0 * 16 + nf) * 32 + lane) * 16);
        v16bf b1 = *(const v16bf*)(bg + ((1 * 16 + nf) * 32 + lane) * 16);
        acc[nf] = __builtin_amdgcn_wmma_f32_16x16x32_bf16(
                      false, afrag[0].v, false, b0, (short)0, acc[nf], false, false);
        acc[nf] = __builtin_amdgcn_wmma_f32_16x16x32_bf16(
                      false, afrag[1].v, false, b1, (short)0, acc[nf], false, false);
    }
    __syncthreads();   // zeros done everywhere; con now in registers

    // C-layout addressing: acc[nf][r] holds con[m = r + 8h, f = 16*nf + fo]
    #define IDX(r, nf) (((r) + 8 * h) * RS + (nf) * 16 + fo)

    // ---- t = sum_g con ------------------------------------------------------
    #pragma unroll
    for (int nf = 0; nf < 16; ++nf)
        #pragma unroll
        for (int r = 0; r < 8; ++r)
            atomicAdd(&bufA[IDX(r, nf)], acc[nf][r]);
    __syncthreads();

    // ---- beta1_g = 0.5 * <con_g, t>_f ; alpha1 = sigmoid -------------------
    float beta1[8], alpha1[8];
    #pragma unroll
    for (int r = 0; r < 8; ++r) {
        float d = 0.0f;
        #pragma unroll
        for (int nf = 0; nf < 16; ++nf)
            d += acc[nf][r] * bufA[IDX(r, nf)];
        d += __shfl_xor(d, 1, 16);
        d += __shfl_xor(d, 2, 16);
        d += __shfl_xor(d, 4, 16);
        d += __shfl_xor(d, 8, 16);          // full sum over f for pixel m=r+8h
        beta1[r]  = 0.5f * d;
        alpha1[r] = sigmoidf(beta1[r]);
    }

    // ---- v1 = sum_g alpha1_g * con_g ---------------------------------------
    #pragma unroll
    for (int nf = 0; nf < 16; ++nf)
        #pragma unroll
        for (int r = 0; r < 8; ++r)
            atomicAdd(&bufB[IDX(r, nf)], alpha1[r] * acc[nf][r]);
    __syncthreads();

    // ---- beta2_g = beta1_g + <con_g, v1>_f ; alpha2 = sigmoid --------------
    float alpha2[8];
    #pragma unroll
    for (int r = 0; r < 8; ++r) {
        float d = 0.0f;
        #pragma unroll
        for (int nf = 0; nf < 16; ++nf)
            d += acc[nf][r] * bufB[IDX(r, nf)];
        d += __shfl_xor(d, 1, 16);
        d += __shfl_xor(d, 2, 16);
        d += __shfl_xor(d, 4, 16);
        d += __shfl_xor(d, 8, 16);
        alpha2[r] = sigmoidf(beta1[r] + d);
    }

    // ---- out = sum_g alpha2_g * con_g + bias (reuse bufA) ------------------
    for (int i = tid; i < ROWS * RS; i += 256) bufA[i] = 0.0f;
    __syncthreads();
    #pragma unroll
    for (int nf = 0; nf < 16; ++nf)
        #pragma unroll
        for (int r = 0; r < 8; ++r)
            atomicAdd(&bufA[IDX(r, nf)], alpha2[r] * acc[nf][r]);
    __syncthreads();

    // coalesced float4 writeback with bias
    const float4* bias4 = (const float4*)bias;
    for (int q = tid; q < ROWS * 64; q += 256) {
        int m  = q >> 6;
        int f4 = q & 63;
        float4 a = *(const float4*)&bufA[m * RS + f4 * 4];
        float4 b = bias4[f4];
        float4 o4 = { a.x + b.x, a.y + b.y, a.z + b.z, a.w + b.w };
        *(float4*)&out[(pixBase + m) * 256 + f4 * 4] = o4;
    }
    #undef IDX
}

// ---------------------------------------------------------------------------
// Host launcher
// ---------------------------------------------------------------------------
extern "C" void kernel_launch(void* const* d_in, const int* in_sizes, int n_in,
                              void* d_out, int out_size, void* d_ws, size_t ws_size,
                              hipStream_t stream) {
    const float* x    = (const float*)d_in[0];   // [32,28,28,512] f32
    const float* kern = (const float*)d_in[1];   // [8,64,256] f32
    const float* bias = (const float*)d_in[2];   // [256] f32
    float* out        = (float*)d_out;           // [32,28,28,256] f32
    unsigned short* bfrag = (unsigned short*)d_ws;  // 256 KB bf16 B-fragments

    const int npix = in_sizes[0] / 512;          // 25088
    const int nBfragElems = 8 * 64 * 256;        // 131072

    repack_kernels_bf16<<<nBfragElems / 256, 256, 0, stream>>>(kern, bfrag);
    routed_group_gemm<<<npix / 16, 256, 0, stream>>>(x, bfrag, bias, out);
}
